// COIL_21071109554433
// MI455X (gfx1250) — matrix operand order repaired and checked by reading, not compiled, not run on hardware
//
#include <hip/hip_runtime.h>

// ---------------------------------------------------------------------------
// COIL scoring + softmax-CE loss for MI455X (gfx1250), bf16 WMMA pipeline
// with Tensor Data Mover (TDM) staging of the interaction tiles into LDS.
// Problem dims (fixed by the reference):
#define QN 32     // queries
#define LQ 32     // query length
#define DN 256    // docs
#define LD 192    // doc length
#define HH 768    // hidden
#define DTOK 128  // token rep dim
#define DCLS 768  // cls rep dim
// ---------------------------------------------------------------------------

typedef __attribute__((ext_vector_type(16))) __bf16        v16bf;
typedef __attribute__((ext_vector_type(8)))  __bf16        v8bf;
typedef __attribute__((ext_vector_type(8)))  float         v8f;
typedef __attribute__((ext_vector_type(4)))  unsigned int  u32x4;
typedef __attribute__((ext_vector_type(8)))  int           i32x8;
typedef __attribute__((ext_vector_type(4)))  int           i32x4;

#if defined(__gfx1250__) && __has_builtin(__builtin_amdgcn_tensor_load_to_lds)
#define HAVE_TDM 1
#endif

// Dynamic-LDS layout for tok_score_kernel (TDM path). Rows are padded by the
// TDM pad feature: 64 DWORDs (256B row) + 4 DWORDs pad -> 272B row stride
// (68 banks -> consecutive rows shift 4 banks; 16-lane fragment reads are
// conflict-free).
#define ROWB    136                  // padded row stride in bf16 elems (272B)
#define SM_DOC  0                    // 192 rows * 272B = 52224
#define SM_QRY  52224                // 32 rows * 272B = 8704
#define SM_QID  60928                // 32 ints = 128B
#define SM_RED  61056                // 32*4 floats = 512B
#define SM_BYTES 61568

__device__ __forceinline__ v8f zero_v8f() {
  v8f z;
#pragma unroll
  for (int i = 0; i < 8; ++i) z[i] = 0.f;
  return z;
}

__device__ __forceinline__ v8f wmma_bf16(v16bf a, v16bf b, v8f c) {
  // D = A(16x32 bf16) * B(32x16 bf16) + C(16x16 f32)
  return __builtin_amdgcn_wmma_f32_16x16x32_bf16(
      /*neg_a=*/false, a, /*neg_b=*/false, b,
      /*c_mod=*/(short)0, c, /*reuse_a=*/false, /*reuse_b=*/false);
}

// A fragment (16x32, 16-bit), contiguous K order (matches B fragments):
//   lanes 0-15 : K = k0+{0..15} in elems 0..15
//   lanes 16-31: K = k0+{16..31}
// pa points at A[row] + k0 (row-major, row = lane&15). hf = lane>>4.
__device__ __forceinline__ v16bf a_frag_from_bf16(const __bf16* pa, int hf) {
  v8bf lo = *(const v8bf*)(pa + hf * 16);
  v8bf hi = *(const v8bf*)(pa + hf * 16 + 8);
  v16bf a;
#pragma unroll
  for (int i = 0; i < 8; ++i) { a[i] = lo[i]; a[i + 8] = hi[i]; }
  return a;
}

// B fragment: 16 contiguous K values (32B) loaded as two 16B chunks (padded
// LDS rows are only 16B aligned).
__device__ __forceinline__ v16bf b_frag16(const __bf16* p) {
  v8bf lo = *(const v8bf*)p;
  v8bf hi = *(const v8bf*)(p + 8);
  v16bf b;
#pragma unroll
  for (int i = 0; i < 8; ++i) { b[i] = lo[i]; b[i + 8] = hi[i]; }
  return b;
}

__device__ __forceinline__ v16bf a_frag_from_f32(const float* pa, int hf) {
  const float4* p0 = (const float4*)(pa + hf * 16);
  const float4* p1 = (const float4*)(pa + hf * 16 + 8);
  float4 f0 = p0[0], f1 = p0[1], f2 = p1[0], f3 = p1[1];
  v16bf a;
  a[0]  = (__bf16)f0.x; a[1]  = (__bf16)f0.y; a[2]  = (__bf16)f0.z; a[3]  = (__bf16)f0.w;
  a[4]  = (__bf16)f1.x; a[5]  = (__bf16)f1.y; a[6]  = (__bf16)f1.z; a[7]  = (__bf16)f1.w;
  a[8]  = (__bf16)f2.x; a[9]  = (__bf16)f2.y; a[10] = (__bf16)f2.z; a[11] = (__bf16)f2.w;
  a[12] = (__bf16)f3.x; a[13] = (__bf16)f3.y; a[14] = (__bf16)f3.z; a[15] = (__bf16)f3.w;
  return a;
}

#ifdef HAVE_TDM
// Issue one TDM load of a [rows x 128] bf16 tile (row stride 128 elems in
// global) into LDS at lds_off, padding each 256B row with 16B.
// D# per CDNA5 ISA. Executed by one wave.
// This toolchain exposes the 6-arg builtin:
//   (u32x4 g0, i32x8 g1, i32x4 g2, i32x4 g3, i32x8 g_ext, i32 cpol)
__device__ __forceinline__ void tdm_load_tile(const __bf16* gptr,
                                              unsigned lds_off, int rows) {
  unsigned long long ga = (unsigned long long)(uintptr_t)gptr;
  u32x4 g0 = {
      1u,                                   // count=1, user mode, gather off
      lds_off,                              // lds_addr (bytes)
      (unsigned)ga,                         // global_addr[31:0]
      (unsigned)((ga >> 32) & 0x1FFFFFFu)   // global_addr[56:32]
          | (2u << 30)                      // type = 2 ("image")
  };
  i32x8 g1 = {
      (int)((1u << 16)       // data_size = 1 -> 2 bytes
            | (1u << 20)     // pad_enable
            | (5u << 22)     // pad_interval: 64 DWORDs per row
            | (3u << 25)),   // pad_amount: 4 DWORDs (16B)
      (int)(DTOK << 16),     // [63:48] tensor_dim0 lo = 128
      (int)(rows << 16),     // [79:64] tdim0 hi = 0, [95:80] tensor_dim1 = rows
      (int)(DTOK << 16),     // [111:96] tdim1 hi = 0, [127:112] tile_dim0 = 128
      rows,                  // [143:128] tile_dim1 = rows, [159:144] tile_dim2=0
      DTOK,                  // [191:160] tensor_dim0_stride = 128
      0,                     // [207:192] stride hi, [223:208] dim1_stride lo
      0                      // [255:224] dim1_stride hi (unused, 2D tile)
  };
  i32x4 gz4 = {0, 0, 0, 0};
  i32x8 gz8 = {0, 0, 0, 0, 0, 0, 0, 0};
  __builtin_amdgcn_tensor_load_to_lds(g0, g1, gz4, gz4, gz8, 0);
}
#endif

// ---------------------------------------------------------------------------
// Transpose + downconvert weights: Wt[n*K + k] = (bf16)W[k*N + n]
// ---------------------------------------------------------------------------
__global__ __launch_bounds__(256)
void transpose_w_kernel(const float* __restrict__ W, __bf16* __restrict__ Wt,
                        int K, int N) {
  int idx = blockIdx.x * 256 + threadIdx.x;
  if (idx < K * N) {
    int n = idx / K, k = idx - n * K;
    Wt[idx] = (__bf16)W[(size_t)k * N + n];
  }
}

// ---------------------------------------------------------------------------
// C[M,Ntot](bf16) = act( A[M,K=768](f32) @ Bt[Ntot,K](bf16)^T + bias )
// Block = 256 thr (8 waves). Tile: 64 rows x 128 cols. Wave owns 16x64.
// grid.x = ceil(M/64), grid.y = Ntot/128.
// ---------------------------------------------------------------------------
__global__ __launch_bounds__(256)
void proj_gemm_kernel(const float* __restrict__ A, int lda, int M,
                      const __bf16* __restrict__ Bt, int Ntot,
                      const float* __restrict__ bias, int do_relu,
                      __bf16* __restrict__ C) {
  const int K = HH;
  int m0  = blockIdx.x * 64;
  int nb0 = blockIdx.y * 128;
  int tid = threadIdx.x;
  int w = tid >> 5, lane = tid & 31;
  int mt = w & 3;          // m tile 0..3
  int nh = w >> 2;         // n half 0..1 -> tiles nh*4 .. nh*4+3
  int hf = lane >> 4, cl = lane & 15;

  int rowA = m0 + mt * 16 + cl;
  int rowC = rowA < M ? rowA : (M - 1);  // clamp; guarded on store
  const float* pa = A + (size_t)rowC * lda;

  v8f acc[4];
#pragma unroll
  for (int t = 0; t < 4; ++t) acc[t] = zero_v8f();

  for (int k0 = 0; k0 < K; k0 += 32) {
    v16bf a = a_frag_from_f32(pa + k0, hf);
#pragma unroll
    for (int t = 0; t < 4; ++t) {
      int nrow = nb0 + (nh * 4 + t) * 16 + cl;
      v16bf b = *(const v16bf*)(Bt + (size_t)nrow * K + k0 + hf * 16);
      acc[t] = wmma_bf16(a, b, acc[t]);
    }
  }

#pragma unroll
  for (int t = 0; t < 4; ++t) {
    int colO = nb0 + (nh * 4 + t) * 16 + cl;
    float bv = bias[colO];
#pragma unroll
    for (int r = 0; r < 8; ++r) {
      int row = m0 + mt * 16 + hf * 8 + r;  // C layout: lanes0-15 M=r, 16-31 M=8+r
      if (row < M) {
        float v = acc[t][r] + bv;
        if (do_relu) v = fmaxf(v, 0.f);
        C[(size_t)row * Ntot + colO] = (__bf16)v;
      }
    }
  }
}

// ---------------------------------------------------------------------------
// cls_scores[QN,DN] = qry_cls(bf16)[QN,DCLS] @ doc_cls(bf16)[DN,DCLS]^T
// grid = (DN/16, QN/16), 32 threads (1 wave), one 16x16 tile each.
// ---------------------------------------------------------------------------
__global__ __launch_bounds__(32)
void cls_score_kernel(const __bf16* __restrict__ qc, const __bf16* __restrict__ dc,
                      float* __restrict__ out) {
  int lane = threadIdx.x & 31;
  int hf = lane >> 4, cl = lane & 15;
  int n0 = blockIdx.x * 16, m0 = blockIdx.y * 16;
  const __bf16* pa = qc + (size_t)(m0 + cl) * DCLS;
  const __bf16* pb = dc + (size_t)(n0 + cl) * DCLS;
  v8f acc = zero_v8f();
  for (int k0 = 0; k0 < DCLS; k0 += 32) {
    v16bf a = a_frag_from_bf16(pa + k0, hf);
    v16bf b = *(const v16bf*)(pb + k0 + hf * 16);
    acc = wmma_bf16(a, b, acc);
  }
#pragma unroll
  for (int r = 0; r < 8; ++r)
    out[(size_t)(m0 + hf * 8 + r) * DN + n0 + cl] = acc[r];
}

// ---------------------------------------------------------------------------
// COIL interaction for one (q,d) pair per block:
//   S[i,j] = qry_reps[q,i,:] . doc_reps[d,j,:]   (32x192, K=128, bf16 WMMA)
//   pooled[i] = max_j (S[i,j] * (qid[i]==did[j]))
//   tok = sum_{i>=1} pooled[i] * qmask[q,i] ;  out = tok + cls_scores[q,d]
// 256 thr = 8 waves; wave (mt = w&1, ng = w>>1) owns rows 16*mt..+16,
// cols 48*ng..+48 (3 WMMA tiles). Tiles staged in LDS via the TDM when
// available (one wave issues tensor_load_to_lds for doc+qry tiles).
// ---------------------------------------------------------------------------
__global__ __launch_bounds__(256)
void tok_score_kernel(const __bf16* __restrict__ qreps, const __bf16* __restrict__ dreps,
                      const int* __restrict__ qids, const int* __restrict__ dids,
                      const float* __restrict__ qmask, const float* __restrict__ clssc,
                      float* __restrict__ out) {
  extern __shared__ char smem_raw[];
  int d = blockIdx.x, q = blockIdx.y;
  int tid = threadIdx.x;
  int w = tid >> 5, lane = tid & 31;
  int mt = w & 1, ng = w >> 1;
  int hf = lane >> 4, cl = lane & 15;

  int*   s_qid         = (int*)(smem_raw + SM_QID);
  float (*s_red)[4]    = (float(*)[4])(smem_raw + SM_RED);

  if (tid < LQ) s_qid[tid] = qids[q * LQ + tid];

  v8f acc0 = zero_v8f(), acc1 = zero_v8f(), acc2 = zero_v8f();

#ifdef HAVE_TDM
  // Async DMA: whole doc tile (192x128 bf16) + qry tile (32x128 bf16) to LDS.
  if (w == 0) {
    tdm_load_tile(dreps + (size_t)d * LD * DTOK, SM_DOC, LD);
    tdm_load_tile(qreps + (size_t)q * LQ * DTOK, SM_QRY, LQ);
    __builtin_amdgcn_s_wait_tensorcnt(0);
  }
  __syncthreads();

  const __bf16* sDoc = (const __bf16*)(smem_raw + SM_DOC);
  const __bf16* sQry = (const __bf16*)(smem_raw + SM_QRY);
  const __bf16* Arow = sQry + (size_t)(mt * 16 + cl) * ROWB;
  const __bf16* B0   = sDoc + (size_t)(ng * 48 + cl) * ROWB;
  const __bf16* B1   = B0 + 16 * ROWB;
  const __bf16* B2   = B0 + 32 * ROWB;
#pragma unroll
  for (int kk = 0; kk < 4; ++kk) {
    int k0 = kk * 32;
    v16bf a  = a_frag_from_bf16(Arow + k0, hf);
    v16bf b0 = b_frag16(B0 + k0 + hf * 16);
    v16bf b1 = b_frag16(B1 + k0 + hf * 16);
    v16bf b2 = b_frag16(B2 + k0 + hf * 16);
    acc0 = wmma_bf16(a, b0, acc0);
    acc1 = wmma_bf16(a, b1, acc1);
    acc2 = wmma_bf16(a, b2, acc2);
  }
#else
  __syncthreads();
  const __bf16* Aq = qreps + ((size_t)q * LQ + mt * 16 + cl) * DTOK;
  const __bf16* B0 = dreps + ((size_t)d * LD + ng * 48 + cl) * DTOK;
  const __bf16* B1 = B0 + 16 * DTOK;
  const __bf16* B2 = B0 + 32 * DTOK;
#pragma unroll
  for (int kk = 0; kk < 4; ++kk) {
    int k0 = kk * 32;
    v16bf a  = a_frag_from_bf16(Aq + k0, hf);
    v16bf b0 = *(const v16bf*)(B0 + k0 + hf * 16);
    v16bf b1 = *(const v16bf*)(B1 + k0 + hf * 16);
    v16bf b2 = *(const v16bf*)(B2 + k0 + hf * 16);
    acc0 = wmma_bf16(a, b0, acc0);
    acc1 = wmma_bf16(a, b1, acc1);
    acc2 = wmma_bf16(a, b2, acc2);
  }
#endif

  // doc token ids for this lane's three j columns
  int jb = d * LD + ng * 48 + cl;
  int dj0 = dids[jb], dj1 = dids[jb + 16], dj2 = dids[jb + 32];

  // Exact-match mask + max over j (lanes within each 16-lane half share i).
#pragma unroll
  for (int r = 0; r < 8; ++r) {
    int i = mt * 16 + hf * 8 + r;
    int qid = s_qid[i];
    float v0 = (qid == dj0) ? acc0[r] : 0.f;
    float v1 = (qid == dj1) ? acc1[r] : 0.f;
    float v2 = (qid == dj2) ? acc2[r] : 0.f;
    float v = fmaxf(v0, fmaxf(v1, v2));
    v = fmaxf(v, __shfl_xor(v, 1, 32));
    v = fmaxf(v, __shfl_xor(v, 2, 32));
    v = fmaxf(v, __shfl_xor(v, 4, 32));
    v = fmaxf(v, __shfl_xor(v, 8, 32));
    if (cl == 0) s_red[i][ng] = v;  // lanes 0 and 16 each publish their half
  }
  __syncthreads();

  if (tid < LQ) {
    int i = tid;
    float p = fmaxf(fmaxf(s_red[i][0], s_red[i][1]),
                    fmaxf(s_red[i][2], s_red[i][3]));
    float term = (i >= 1) ? p * qmask[q * LQ + i] : 0.f;
    term += __shfl_xor(term, 1, 32);
    term += __shfl_xor(term, 2, 32);
    term += __shfl_xor(term, 4, 32);
    term += __shfl_xor(term, 8, 32);
    term += __shfl_xor(term, 16, 32);
    if (tid == 0) out[q * DN + d] = term + clssc[q * DN + d];
  }
}

// ---------------------------------------------------------------------------
// Softmax cross-entropy: loss = mean_q -log_softmax(scores[q])[q*group]
// ---------------------------------------------------------------------------
__global__ __launch_bounds__(256)
void loss_kernel(const float* __restrict__ scores, const int* __restrict__ grp,
                 float* __restrict__ out_loss) {
  __shared__ float red[256];
  int t = threadIdx.x;
  int g = grp[0];
  float lsum = 0.f;
  for (int q = 0; q < QN; ++q) {
    float v = scores[q * DN + t];
    red[t] = v;
    __syncthreads();
    for (int s = 128; s > 0; s >>= 1) {
      if (t < s) red[t] = fmaxf(red[t], red[t + s]);
      __syncthreads();
    }
    float mx = red[0];
    __syncthreads();
    red[t] = __expf(v - mx);
    __syncthreads();
    for (int s = 128; s > 0; s >>= 1) {
      if (t < s) red[t] += red[t + s];
      __syncthreads();
    }
    if (t == 0) {
      float lse = mx + __logf(red[0]);
      lsum += -(scores[q * DN + q * g] - lse);
    }
    __syncthreads();
  }
  if (t == 0) out_loss[0] = lsum / (float)QN;
}

// ---------------------------------------------------------------------------
extern "C" void kernel_launch(void* const* d_in, const int* in_sizes, int n_in,
                              void* d_out, int out_size, void* d_ws, size_t ws_size,
                              hipStream_t stream) {
  (void)in_sizes; (void)n_in; (void)out_size; (void)ws_size;

  const float* qry_hidden = (const float*)d_in[0];  // [32,32,768]
  const float* doc_hidden = (const float*)d_in[1];  // [256,192,768]
  const int*   qry_ids    = (const int*)  d_in[2];  // [32,32]
  const int*   doc_ids    = (const int*)  d_in[3];  // [256,192]
  const float* qmask      = (const float*)d_in[4];  // [32,32]
  const float* tok_w      = (const float*)d_in[5];  // [768,128]
  const float* tok_b      = (const float*)d_in[6];  // [128]
  const float* cls_w      = (const float*)d_in[7];  // [768,768]
  const float* cls_b      = (const float*)d_in[8];  // [768]
  const int*   group      = (const int*)  d_in[9];  // scalar (8)

  // Workspace layout (all offsets 256B aligned; total ~14.7 MB)
  char* ws = (char*)d_ws;
  __bf16* tok_wt   = (__bf16*)(ws + 0);         // [128][768]   196608 B
  __bf16* cls_wt   = (__bf16*)(ws + 196608);    // [768][768]  1179648 B
  __bf16* doc_reps = (__bf16*)(ws + 1376256);   // [256*192][128] 12582912 B
  __bf16* qry_reps = (__bf16*)(ws + 13959168);  // [32*32][128]  262144 B
  __bf16* qry_cls  = (__bf16*)(ws + 14221312);  // [32][768]      49152 B
  __bf16* doc_cls  = (__bf16*)(ws + 14270464);  // [256][768]    393216 B
  float*  cls_sc   = (float*) (ws + 14663680);  // [32][256]      32768 B

  float* out    = (float*)d_out;  // [0]=loss, [1..8192]=scores
  float* scores = out + 1;

  // 1) transpose + bf16 weights
  transpose_w_kernel<<<dim3((HH * DTOK + 255) / 256), 256, 0, stream>>>(
      tok_w, tok_wt, HH, DTOK);
  transpose_w_kernel<<<dim3((HH * DCLS + 255) / 256), 256, 0, stream>>>(
      cls_w, cls_wt, HH, DCLS);

  // 2) token projections (relu, bf16 out)
  proj_gemm_kernel<<<dim3((DN * LD) / 64, 1), 256, 0, stream>>>(
      doc_hidden, HH, DN * LD, tok_wt, DTOK, tok_b, 1, doc_reps);
  proj_gemm_kernel<<<dim3((QN * LQ) / 64, 1), 256, 0, stream>>>(
      qry_hidden, HH, QN * LQ, tok_wt, DTOK, tok_b, 1, qry_reps);

  // 3) cls projections (rows = position 0 of each sequence; lda = seq stride)
  proj_gemm_kernel<<<dim3(1, DCLS / 128), 256, 0, stream>>>(
      qry_hidden, LQ * HH, QN, cls_wt, DCLS, cls_b, 0, qry_cls);
  proj_gemm_kernel<<<dim3(DN / 64, DCLS / 128), 256, 0, stream>>>(
      doc_hidden, LD * HH, DN, cls_wt, DCLS, cls_b, 0, doc_cls);

  // 4) cls dot-product scores
  cls_score_kernel<<<dim3(DN / 16, QN / 16), 32, 0, stream>>>(
      qry_cls, doc_cls, cls_sc);

  // 5) COIL interaction (TDM-staged tiles): masked max-pool + sum + cls add.
  //    Dynamic LDS size passed unconditionally so host/device passes agree.
  tok_score_kernel<<<dim3(DN, QN), 256, SM_BYTES, stream>>>(
      qry_reps, doc_reps, qry_ids, doc_ids, qmask, cls_sc, scores);

  // 6) softmax cross-entropy loss
  loss_kernel<<<1, 256, 0, stream>>>(scores, group, out);
}
